// MambaEncoder_73083163509406
// MI455X (gfx1250) — matrix operand order repaired and checked by reading, not compiled, not run on hardware
//
#include <hip/hip_runtime.h>
#include <hip/hip_bf16.h>
#include <math.h>

// ---------------- model constants ----------------
#define Bq      4
#define Tq      2048
#define BT      (Bq * Tq)          // 8192 tokens
#define IN_DIM  256
#define DM      512                 // D_MODEL
#define DI      1024                // D_INNER
#define DS      16                  // DSTATE
#define NH      8                   // NHEADS
#define HD      128                 // HEADDIM
#define CD      1056                // CONV_DIM
#define DIP     2088                // D_IN_PROJ
#define LDZ     2096                // padded stride for zxbcdt buffer
#define DIPPAD  2112                // DIP padded to multiple of 64 (weight transpose)
#define NL      4
#define EPSF    1e-5f

typedef __attribute__((ext_vector_type(16))) __bf16 v16bf;
typedef __attribute__((ext_vector_type(8)))  __bf16 v8bf;
typedef __attribute__((ext_vector_type(8)))  float  v8f;

// ---------------- fp32 -> bf16 convert ----------------
__global__ void tobf16_kernel(const float* __restrict__ src, __bf16* __restrict__ dst, size_t n) {
    size_t i = (size_t)blockIdx.x * blockDim.x + threadIdx.x;
    if (i < n) dst[i] = (__bf16)src[i];
}

// ---------------- weight transpose+pad: wt[k*Npad+n] = (n<N) ? w[n*K+k] : 0 ----------------
__global__ void transpose_pad_bf16(const float* __restrict__ w, __bf16* __restrict__ wt,
                                   int N, int K, int Npad) {
    int idx = blockIdx.x * blockDim.x + threadIdx.x;
    if (idx >= K * Npad) return;
    int n = idx % Npad, k = idx / Npad;
    wt[idx] = (n < N) ? (__bf16)w[(size_t)n * K + k] : (__bf16)0.f;
}

// ---------------- WMMA GEMM: C[m,n] = sum_k A[m,k] * Wt[k,n] (+bias[n]) (+resid) ----------------
// A : (M,K) bf16 row-major.   Wt : (K,Npad) bf16 row-major, zero-padded columns.
// One wave computes a 32x64 macro-tile = 2x4 WMMA accumulators; K stepped by 32.
// 8 WMMAs per 4 b128 A loads + 64 strided bf16 B loads, no guards inside the K-loop (EXEC stays all-ones).
__global__ __launch_bounds__(256)
void gemm_bf16_wmma(const __bf16* __restrict__ A, const __bf16* __restrict__ Wt,
                    const float* __restrict__ bias, const float* __restrict__ resid,
                    float* __restrict__ Cf, __bf16* __restrict__ Cbf,
                    int M, int N, int K, int ldc, int Npad, int wtiles_n, int total_wtiles) {
    int wave = (int)((blockIdx.x * blockDim.x + threadIdx.x) >> 5);
    if (wave >= total_wtiles) return;               // uniform per wave
    int lane = threadIdx.x & 31;
    int wm = wave / wtiles_n, wn = wave % wtiles_n;
    int mlo = lane & 15, hi = lane >> 4;
    int m0 = wm * 32, n0 = wn * 64;
    const __bf16* arow0 = A + (size_t)(m0 + mlo) * K;
    const __bf16* arow1 = arow0 + (size_t)16 * K;

    v8f acc[2][4];
#pragma unroll
    for (int tni = 0; tni < 4; ++tni) {
        float bv = 0.f;
        if (bias) {
            int gn = n0 + tni * 16 + mlo;
            bv = (gn < N) ? bias[gn] : 0.f;
        }
#pragma unroll
        for (int r = 0; r < 8; ++r) { acc[0][tni][r] = bv; acc[1][tni][r] = bv; }
    }

    for (int k0 = 0; k0 < K; k0 += 32) {
        // A fragments: element j<8 -> k=k0+hi*8+j ; j>=8 -> k=k0+16+hi*8+(j-8)
        v8bf a0lo = *(const v8bf*)(arow0 + k0 + hi * 8);
        v8bf a0up = *(const v8bf*)(arow0 + k0 + 16 + hi * 8);
        v8bf a1lo = *(const v8bf*)(arow1 + k0 + hi * 8);
        v8bf a1up = *(const v8bf*)(arow1 + k0 + 16 + hi * 8);
        v16bf a0 = __builtin_shufflevector(a0lo, a0up, 0,1,2,3,4,5,6,7,8,9,10,11,12,13,14,15);
        v16bf a1 = __builtin_shufflevector(a1lo, a1up, 0,1,2,3,4,5,6,7,8,9,10,11,12,13,14,15);
        // B fragments: lane -> n = n0 + tni*16 + (lane&15), elements j -> k same as A (j<8: hi*8+j; j>=8: 16+hi*8+(j-8))
        const __bf16* brow = Wt + (size_t)k0 * Npad + n0 + mlo;
        v16bf bfr[4];
#pragma unroll
        for (int tni = 0; tni < 4; ++tni)
#pragma unroll
            for (int j = 0; j < 16; ++j)
                bfr[tni][j] = brow[(size_t)((j < 8) ? (hi * 8 + j) : (16 + hi * 8 + (j - 8))) * Npad + tni * 16];
#pragma unroll
        for (int tni = 0; tni < 4; ++tni) {
            acc[0][tni] = __builtin_amdgcn_wmma_f32_16x16x32_bf16(false, a0, false, bfr[tni],
                                                                  (short)0, acc[0][tni], false, false);
            acc[1][tni] = __builtin_amdgcn_wmma_f32_16x16x32_bf16(false, a1, false, bfr[tni],
                                                                  (short)0, acc[1][tni], false, false);
        }
    }

    // epilogue: C/D layout -> lane n = lane&15, row m = hi*8 + r
#pragma unroll
    for (int tmi = 0; tmi < 2; ++tmi) {
#pragma unroll
        for (int tni = 0; tni < 4; ++tni) {
            int gn = n0 + tni * 16 + mlo;
            if (gn < N) {
#pragma unroll
                for (int r = 0; r < 8; ++r) {
                    int gm = m0 + tmi * 16 + hi * 8 + r;
                    float v = acc[tmi][tni][r];
                    if (resid) v += resid[(size_t)gm * ldc + gn];
                    Cf[(size_t)gm * ldc + gn] = v;
                    if (Cbf) Cbf[(size_t)gm * ldc + gn] = (__bf16)v;
                }
            }
        }
    }
}

// ---------------- LayerNorm over last dim D (D = 512 or 1024, block = 256) ----------------
__global__ void layernorm_kernel(const float* __restrict__ x, const float* __restrict__ w,
                                 const float* __restrict__ b, float* __restrict__ y,
                                 __bf16* __restrict__ ybf, int D) {
    int bt = blockIdx.x, tid = threadIdx.x;
    __shared__ float red[256];
    const float* row = x + (size_t)bt * D;
    int cnt = D / 256;
    float vals[4];
    float s = 0.f;
    for (int i = 0; i < cnt; ++i) { vals[i] = row[tid + i * 256]; s += vals[i]; }
    red[tid] = s; __syncthreads();
    for (int st = 128; st > 0; st >>= 1) { if (tid < st) red[tid] += red[tid + st]; __syncthreads(); }
    float mean = red[0] / (float)D; __syncthreads();
    float v = 0.f;
    for (int i = 0; i < cnt; ++i) { float d = vals[i] - mean; v += d * d; }
    red[tid] = v; __syncthreads();
    for (int st = 128; st > 0; st >>= 1) { if (tid < st) red[tid] += red[tid + st]; __syncthreads(); }
    float inv = rsqrtf(red[0] / (float)D + EPSF);
    for (int i = 0; i < cnt; ++i) {
        int j = tid + i * 256;
        float o = (vals[i] - mean) * inv * w[j] + b[j];
        y[(size_t)bt * D + j] = o;
        if (ybf) ybf[(size_t)bt * D + j] = (__bf16)o;
    }
}

// ---------------- time-reverse copy (per batch) ----------------
__global__ void reverse_kernel(const float* __restrict__ src, float* __restrict__ dst,
                               __bf16* __restrict__ dstbf, int T, int D) {
    int bt = blockIdx.x; int bb = bt / T, t = bt % T;
    const float* s = src + (size_t)(bb * T + (T - 1 - t)) * D;
    float* d = dst + (size_t)bt * D;
    __bf16* db = dstbf + (size_t)bt * D;
    for (int j = threadIdx.x; j < D; j += blockDim.x) { float v = s[j]; d[j] = v; db[j] = (__bf16)v; }
}

// ---------------- causal depthwise conv (DCONV=4) + bias + SiLU ----------------
__global__ void conv_silu_kernel(const float* __restrict__ zx, const float* __restrict__ cw,
                                 const float* __restrict__ cb, float* __restrict__ xconv, int T) {
    int e = blockIdx.x * blockDim.x + threadIdx.x;
    if (e >= BT * CD) return;
    int c = e % CD; int bt = e / CD;
    int bb = bt / T, t = bt % T;
    float acc = cb[c];
#pragma unroll
    for (int j = 0; j < 4; ++j) {
        int tt = t - 3 + j;
        if (tt >= 0) acc += zx[(size_t)(bb * T + tt) * LDZ + DI + c] * cw[c * 4 + j];
    }
    acc = acc / (1.f + __expf(-acc));               // SiLU
    xconv[(size_t)bt * CD + c] = acc;
}

// ---------------- dt = softplus(dt + dt_bias); dA = exp(-exp(A_log)*dt) ----------------
__global__ void dt_kernel(const float* __restrict__ zx, const float* __restrict__ dt_bias,
                          const float* __restrict__ A_log, float* __restrict__ dtv,
                          float* __restrict__ dAv) {
    int e = blockIdx.x * blockDim.x + threadIdx.x;
    if (e >= BT * NH) return;
    int h = e % NH; int bt = e / NH;
    float d = zx[(size_t)bt * LDZ + DI + CD + h] + dt_bias[h];
    float sp = (d > 20.f) ? d : log1pf(__expf(d));
    dtv[e] = sp;
    dAv[e] = __expf(-__expf(A_log[h]) * sp);
}

// ---------------- sequential selective scan: one block per (batch, head) ----------------
__global__ void scan_kernel(const float* __restrict__ xconv, const float* __restrict__ dtv,
                            const float* __restrict__ dAv, const float* __restrict__ Dcoef,
                            float* __restrict__ yout, int T) {
    int bb = blockIdx.x / NH, h = blockIdx.x % NH;
    int p = threadIdx.x;                            // 0..127
    float s[DS];
#pragma unroll
    for (int n = 0; n < DS; ++n) s[n] = 0.f;
    float Dv = Dcoef[h];
    for (int t = 0; t < T; ++t) {
        size_t bt = (size_t)(bb * T + t);
        const float* row = xconv + bt * CD;
        float dA = dAv[bt * NH + h];
        float dt = dtv[bt * NH + h];
        float x  = row[h * HD + p];
        float xdt = x * dt;
        float acc = 0.f;
#pragma unroll
        for (int n = 0; n < DS; ++n) {
            float Bn = row[DI + n];                 // broadcast loads (same addr all lanes)
            float Cn = row[DI + DS + n];
            s[n] = dA * s[n] + xdt * Bn;
            acc += s[n] * Cn;
        }
        yout[bt * DI + h * HD + p] = acc + Dv * x;
    }
}

// ---------------- gated RMSNorm: g = y*silu(z); g *= rsqrt(mean g^2 + eps)*rms_w -> bf16 ----------------
__global__ void gated_rms_kernel(const float* __restrict__ y, const float* __restrict__ zx,
                                 const float* __restrict__ rms_w, __bf16* __restrict__ gbf) {
    int bt = blockIdx.x, tid = threadIdx.x;
    __shared__ float red[256];
    float g[4]; float local = 0.f;
#pragma unroll
    for (int i = 0; i < 4; ++i) {
        int j = tid + i * 256;
        float zv = zx[(size_t)bt * LDZ + j];
        float gv = y[(size_t)bt * DI + j] * (zv / (1.f + __expf(-zv)));
        g[i] = gv; local += gv * gv;
    }
    red[tid] = local; __syncthreads();
    for (int st = 128; st > 0; st >>= 1) { if (tid < st) red[tid] += red[tid + st]; __syncthreads(); }
    float scale = rsqrtf(red[0] / (float)DI + EPSF);
#pragma unroll
    for (int i = 0; i < 4; ++i) {
        int j = tid + i * 256;
        gbf[(size_t)bt * DI + j] = (__bf16)(g[i] * scale * rms_w[j]);
    }
}

// ---------------- final concat(fwd, reversed bwd) + LayerNorm(1024) ----------------
__global__ void concat_ln_kernel(const float* __restrict__ hf, const float* __restrict__ hb,
                                 const float* __restrict__ w, const float* __restrict__ b,
                                 float* __restrict__ out, int T) {
    int bt = blockIdx.x; int bb = bt / T, t = bt % T;
    int tid = threadIdx.x;
    __shared__ float red[256];
    size_t fo = (size_t)bt * DM;
    size_t bo = (size_t)(bb * T + (T - 1 - t)) * DM;
    float vals[4]; float s = 0.f;
#pragma unroll
    for (int i = 0; i < 4; ++i) {
        int j = tid + i * 256;
        float v = (j < DM) ? hf[fo + j] : hb[bo + (j - DM)];
        vals[i] = v; s += v;
    }
    red[tid] = s; __syncthreads();
    for (int st = 128; st > 0; st >>= 1) { if (tid < st) red[tid] += red[tid + st]; __syncthreads(); }
    float mean = red[0] / 1024.f; __syncthreads();
    float v = 0.f;
#pragma unroll
    for (int i = 0; i < 4; ++i) { float d = vals[i] - mean; v += d * d; }
    red[tid] = v; __syncthreads();
    for (int st = 128; st > 0; st >>= 1) { if (tid < st) red[tid] += red[tid + st]; __syncthreads(); }
    float inv = rsqrtf(red[0] / 1024.f + EPSF);
#pragma unroll
    for (int i = 0; i < 4; ++i) {
        int j = tid + i * 256;
        out[(size_t)bt * 1024 + j] = (vals[i] - mean) * inv * w[j] + b[j];
    }
}

// ================== host orchestration ==================
extern "C" void kernel_launch(void* const* d_in, const int* in_sizes, int n_in,
                              void* d_out, int out_size, void* d_ws, size_t ws_size,
                              hipStream_t stream) {
    (void)in_sizes; (void)n_in; (void)out_size; (void)ws_size;
    const float* x        = (const float*)d_in[0];
    const float* embed_w  = (const float*)d_in[1];
    const float* embed_b  = (const float*)d_in[2];
    const float* ln1_w    = (const float*)d_in[3];
    const float* ln1_b    = (const float*)d_in[4];
    const float* lnout_w  = (const float*)d_in[5];
    const float* lnout_b  = (const float*)d_in[6];
    // per-direction param blocks start at 7 (fwd) and 15 (bwd)
    // order: in_w, conv_w, conv_b, dt_bias, A_log, D, rms_w, out_w

    char* base = (char*)d_ws;
    size_t off = 0;
    auto alloc = [&](size_t bytes) { size_t o = off; off += (bytes + 255) & ~(size_t)255; return o; };
    float*  hf    = (float*)(base + alloc((size_t)BT * DM * 4));
    __bf16* hfb   = (__bf16*)(base + alloc((size_t)BT * DM * 2));
    float*  hb    = (float*)(base + alloc((size_t)BT * DM * 4));
    __bf16* hbb   = (__bf16*)(base + alloc((size_t)BT * DM * 2));
    float*  zx    = (float*)(base + alloc((size_t)BT * LDZ * 4));
    float*  xconv = (float*)(base + alloc((size_t)BT * CD * 4));
    float*  dtv   = (float*)(base + alloc((size_t)BT * NH * 4));
    float*  dAv   = (float*)(base + alloc((size_t)BT * NH * 4));
    float*  ybuf  = (float*)(base + alloc((size_t)BT * DI * 4));
    __bf16* gbf   = (__bf16*)(base + alloc((size_t)BT * DI * 2));
    __bf16* xbf   = (__bf16*)(base + alloc((size_t)BT * IN_DIM * 2));
    __bf16* ewt   = (__bf16*)(base + alloc((size_t)IN_DIM * DM * 2));          // (K=256, Npad=512)
    __bf16* inwt  = (__bf16*)(base + alloc((size_t)2 * NL * DM * DIPPAD * 2)); // (K=512, Npad=2112)/layer
    __bf16* outwt = (__bf16*)(base + alloc((size_t)2 * NL * DI * DM * 2));     // (K=1024, Npad=512)/layer

    auto cvt = [&](const float* s, __bf16* d, size_t n) {
        tobf16_kernel<<<(unsigned)((n + 255) / 256), 256, 0, stream>>>(s, d, n);
    };
    auto tpad = [&](const float* w, __bf16* wt, int N, int K, int Npad) {
        int n = K * Npad;
        transpose_pad_bf16<<<(n + 255) / 256, 256, 0, stream>>>(w, wt, N, K, Npad);
    };
    auto gemm = [&](const __bf16* A, const __bf16* Wt, const float* bias, const float* resid,
                    float* Cf, __bf16* Cb, int M, int N, int K, int ldc, int Npad) {
        int wtiles_n = Npad / 64;
        int total = (M / 32) * wtiles_n;
        int blocks = (total + 7) / 8;                 // 8 waves / block
        gemm_bf16_wmma<<<blocks, 256, 0, stream>>>(A, Wt, bias, resid, Cf, Cb,
                                                   M, N, K, ldc, Npad, wtiles_n, total);
    };

    // ---- weight prep: bf16 + transpose + pad ----
    cvt(x, xbf, (size_t)BT * IN_DIM);
    tpad(embed_w, ewt, DM, IN_DIM, DM);
    for (int dir = 0; dir < 2; ++dir) {
        const float* in_w  = (const float*)d_in[dir ? 15 : 7];
        const float* out_w = (const float*)d_in[dir ? 22 : 14];
        for (int i = 0; i < NL; ++i) {
            tpad(in_w  + (size_t)i * DIP * DM, inwt  + ((size_t)dir * NL + i) * DM * DIPPAD, DIP, DM, DIPPAD);
            tpad(out_w + (size_t)i * DM * DI,  outwt + ((size_t)dir * NL + i) * DI * DM,     DM,  DI, DM);
        }
    }

    // ---- embed + bias -> zx (stride DM), then LN -> hf(+bf16), reverse -> hb ----
    gemm(xbf, ewt, embed_b, nullptr, zx, nullptr, BT, DM, IN_DIM, DM, DM);
    layernorm_kernel<<<BT, 256, 0, stream>>>(zx, ln1_w, ln1_b, hf, hfb, DM);
    reverse_kernel<<<BT, 256, 0, stream>>>(hf, hb, hbb, Tq, DM);

    for (int dir = 0; dir < 2; ++dir) {
        float*  h    = dir ? hb : hf;
        __bf16* hbfp = dir ? hbb : hfb;
        int pb = dir ? 15 : 7;
        const float* conv_w  = (const float*)d_in[pb + 1];
        const float* conv_b  = (const float*)d_in[pb + 2];
        const float* dt_bias = (const float*)d_in[pb + 3];
        const float* A_log   = (const float*)d_in[pb + 4];
        const float* Dcoef   = (const float*)d_in[pb + 5];
        const float* rms_w   = (const float*)d_in[pb + 6];
        for (int i = 0; i < NL; ++i) {
            const __bf16* wi = inwt  + ((size_t)dir * NL + i) * DM * DIPPAD;
            const __bf16* wo = outwt + ((size_t)dir * NL + i) * DI * DM;
            gemm(hbfp, wi, nullptr, nullptr, zx, nullptr, BT, DIP, DM, LDZ, DIPPAD);
            conv_silu_kernel<<<(BT * CD + 255) / 256, 256, 0, stream>>>(
                zx, conv_w + (size_t)i * CD * 4, conv_b + (size_t)i * CD, xconv, Tq);
            dt_kernel<<<(BT * NH + 255) / 256, 256, 0, stream>>>(
                zx, dt_bias + (size_t)i * NH, A_log + (size_t)i * NH, dtv, dAv);
            scan_kernel<<<Bq * NH, HD, 0, stream>>>(xconv, dtv, dAv, Dcoef + (size_t)i * NH, ybuf, Tq);
            gated_rms_kernel<<<BT, 256, 0, stream>>>(ybuf, zx, rms_w + (size_t)i * DI, gbf);
            // out_proj + residual; overwrite h (f32) and refresh its bf16 mirror
            gemm(gbf, wo, nullptr, h, h, hbfp, BT, DM, DI, DM, DM);
        }
    }

    concat_ln_kernel<<<BT, 256, 0, stream>>>(hf, hb, lnout_w, lnout_b, (float*)d_out, Tq);
}
